// Transformer_71339406787187
// MI455X (gfx1250) — compile-verified, matrix operand order checked by reading, not run on hardware
//
#include <hip/hip_runtime.h>

// ---------------------------------------------------------------------------
// Types
// ---------------------------------------------------------------------------
typedef __attribute__((ext_vector_type(16))) __bf16 bf16x16;
typedef __attribute__((ext_vector_type(8)))  __bf16 bf16x8;
typedef __attribute__((ext_vector_type(4)))  __bf16 bf16x4;
typedef __attribute__((ext_vector_type(2)))  __bf16 bf16x2;
typedef __attribute__((ext_vector_type(8)))  float  f32x8;
typedef __attribute__((ext_vector_type(4)))  float  f32x4;

#define TFX_B  8
#define TFX_T  1024
#define TFX_C  1024
#define TFX_H  16
#define TFX_HS 64
#define TFX_FF 4096
#define TFX_V  67
#define TFX_L  6

// Native RNE convert (lowers to v_cvt_pk_bf16_f32 when paired)
__device__ __forceinline__ __bf16 f2bf(float f) { return (__bf16)f; }

// CDNA5 async DMA: copy 16 bytes global -> LDS, tracked by ASYNCcnt.
__device__ __forceinline__ void async_copy16(const void* g, void* l) {
  unsigned lds = (unsigned)(uintptr_t)l;                 // LDS byte offset
  unsigned long long ga = (unsigned long long)(uintptr_t)g;
  asm volatile("global_load_async_to_lds_b128 %0, %1, off"
               :: "v"(lds), "v"(ga) : "memory");
}
__device__ __forceinline__ void wait_async0() {
  asm volatile("s_wait_asynccnt 0" ::: "memory");
}

// Load one 16x32 bf16 WMMA operand fragment. Per the CDNA5 ISA layout:
// lane half selects K-offset (0 or 8); element j<8 -> K=koff+j (contiguous),
// element j>=8 -> K=16+koff+(j-8) (contiguous). Two 16-byte loads.
__device__ __forceinline__ bf16x16 ld_frag(const __bf16* p, int koff) {
  bf16x8 lo = *reinterpret_cast<const bf16x8*>(p + koff);
  bf16x8 hi = *reinterpret_cast<const bf16x8*>(p + 16 + koff);
  bf16x16 r;
#pragma unroll
  for (int i = 0; i < 8; ++i) { r[i] = lo[i]; r[i + 8] = hi[i]; }
  return r;
}

__device__ __forceinline__ f32x8 wmma_bf16(bf16x16 a, bf16x16 b, f32x8 c) {
  return __builtin_amdgcn_wmma_f32_16x16x32_bf16(false, a, false, b,
                                                 (short)0, c, false, false);
}

// ---------------------------------------------------------------------------
// Embedding: x[b*T+t, c] = tok_emb[idx[b,t], c] + pos_emb[t, c]
// ---------------------------------------------------------------------------
__global__ __launch_bounds__(256) void embed_kernel(
    const int* __restrict__ idx, const float* __restrict__ tok,
    const float* __restrict__ pos, float* __restrict__ x) {
  const int row = blockIdx.x;                // 0 .. B*T-1
  const int t   = row & (TFX_T - 1);
  const int tid = threadIdx.x;
  const size_t tb = (size_t)idx[row] * TFX_C;
  const size_t pb = (size_t)t * TFX_C;
  const size_t xb = (size_t)row * TFX_C;
#pragma unroll
  for (int c = tid; c < TFX_C; c += 256)
    x[xb + c] = tok[tb + c] + pos[pb + c];
}

// ---------------------------------------------------------------------------
// LayerNorm over C=1024, output bf16 (WMMA A-operand precision)
// ---------------------------------------------------------------------------
__global__ __launch_bounds__(256) void ln_kernel(
    const float* __restrict__ x, const float* __restrict__ g,
    const float* __restrict__ b, __bf16* __restrict__ out) {
  const int row = blockIdx.x;
  const int tid = threadIdx.x;
  const float* xr = x + (size_t)row * TFX_C;
  __shared__ float red[256];

  float s = 0.f;
#pragma unroll
  for (int c = tid; c < TFX_C; c += 256) s += xr[c];
  red[tid] = s; __syncthreads();
  for (int st = 128; st; st >>= 1) {
    if (tid < st) red[tid] += red[tid + st];
    __syncthreads();
  }
  const float mean = red[0] * (1.0f / TFX_C);
  __syncthreads();

  float s2 = 0.f;
#pragma unroll
  for (int c = tid; c < TFX_C; c += 256) {
    float d = xr[c] - mean; s2 += d * d;
  }
  red[tid] = s2; __syncthreads();
  for (int st = 128; st; st >>= 1) {
    if (tid < st) red[tid] += red[tid + st];
    __syncthreads();
  }
  const float rstd = rsqrtf(red[0] * (1.0f / TFX_C) + 1e-5f);

  __bf16* orow = out + (size_t)row * TFX_C;
#pragma unroll
  for (int c = tid * 2; c < TFX_C; c += 512) {
    bf16x2 pk;
    pk[0] = f2bf((xr[c + 0] - mean) * rstd * g[c + 0] + b[c + 0]);
    pk[1] = f2bf((xr[c + 1] - mean) * rstd * g[c + 1] + b[c + 1]);
    *reinterpret_cast<bf16x2*>(orow + c) = pk;
  }
}

// ---------------------------------------------------------------------------
// GEMM: out = act( A[M,K](bf16) x W[K,N](fp32->bf16) + bias + residual )
// 128x128 block tile, K-step 32, 8 waves (2x4), 64x32 per wave.
// Double-buffered LDS; A tile staged with async DMA, W tile via registers
// with packed bf16 converts and b32 transpose stores.
// ---------------------------------------------------------------------------
__global__ __launch_bounds__(256) void gemm_wmma(
    const __bf16* __restrict__ A, const float* __restrict__ W,
    const float* __restrict__ bias, const float* __restrict__ residual,
    float* __restrict__ outF, __bf16* __restrict__ outB,
    int M, int N, int K, int relu) {
  __shared__ alignas(16) __bf16 As[2][128][48];  // [buf][m][k]
  __shared__ alignas(16) __bf16 Bs[2][128][48];  // [buf][n][k] (transposed W)

  const int tid  = threadIdx.x;
  const int lane = tid & 31, wave = tid >> 5;
  const int wm   = wave >> 2, wn = wave & 3;
  const int bm   = blockIdx.x * 128, bn = blockIdx.y * 128;
  const int ln   = lane & 15;
  const int halfIdx = lane >> 4;
  const int koff = halfIdx * 8;
  const bool vecW = ((N & 3) == 0);

  f32x8 acc[4][2];
#pragma unroll
  for (int mt = 0; mt < 4; ++mt)
#pragma unroll
    for (int nt = 0; nt < 2; ++nt)
#pragma unroll
      for (int i = 0; i < 8; ++i) acc[mt][nt][i] = 0.f;

  // W staging registers: two consecutive K rows x 4 columns, x2 chunks
  f32x4 wrA[2], wrB[2];

  auto stageA = [&](int k0, __bf16 (*dst)[48]) {
#pragma unroll
    for (int t = 0; t < 2; ++t) {
      int j = tid + t * 256;                // 0..511
      int r = j >> 2, c8 = (j & 3) << 3;
      async_copy16(A + (size_t)(bm + r) * K + k0 + c8, &dst[r][c8]);
    }
  };
  auto fetchW = [&](int k0) {
#pragma unroll
    for (int t = 0; t < 2; ++t) {
      int j  = tid + t * 256;               // 0..511
      int kp = j >> 5;                      // k-pair index 0..15
      int n4 = (j & 31) << 2;               // 0..124
      const float* wa = W + (size_t)(k0 + 2 * kp + 0) * N + bn + n4;
      const float* wb = W + (size_t)(k0 + 2 * kp + 1) * N + bn + n4;
      if (vecW) {
        wrA[t] = *reinterpret_cast<const f32x4*>(wa);
        wrB[t] = *reinterpret_cast<const f32x4*>(wb);
      } else {
#pragma unroll
        for (int i = 0; i < 4; ++i) {
          wrA[t][i] = (bn + n4 + i < N) ? wa[i] : 0.f;
          wrB[t][i] = (bn + n4 + i < N) ? wb[i] : 0.f;
        }
      }
    }
  };
  auto storeW = [&](__bf16 (*dst)[48]) {
#pragma unroll
    for (int t = 0; t < 2; ++t) {
      int j  = tid + t * 256;
      int kp = j >> 5;
      int n4 = (j & 31) << 2;
#pragma unroll
      for (int i = 0; i < 4; ++i) {
        bf16x2 pk;                          // v_cvt_pk_bf16_f32 + ds_store_b32
        pk[0] = f2bf(wrA[t][i]);
        pk[1] = f2bf(wrB[t][i]);
        *reinterpret_cast<bf16x2*>(&dst[n4 + i][2 * kp]) = pk;
      }
    }
  };

  // --- prologue: stage tile 0 into buffer 0 ---
  stageA(0, As[0]);
  fetchW(0);
  storeW(Bs[0]);
  wait_async0();

  int cur = 0;
  const int ktiles = K >> 5;
  for (int kt = 0; kt < ktiles; ++kt) {
    __syncthreads();                        // buffers [cur] ready for all waves
    const bool more = (kt + 1 < ktiles);
    if (more) {                             // overlap: DMA + global loads for kt+1
      stageA((kt + 1) * 32, As[cur ^ 1]);
      fetchW((kt + 1) * 32);
    }

    bf16x16 af[4];
#pragma unroll
    for (int mt = 0; mt < 4; ++mt)
      af[mt] = ld_frag(&As[cur][wm * 64 + mt * 16 + ln][0], koff);
    bf16x16 bfg[2];
#pragma unroll
    for (int nt = 0; nt < 2; ++nt)
      bfg[nt] = ld_frag(&Bs[cur][wn * 32 + nt * 16 + ln][0], koff);
#pragma unroll
    for (int mt = 0; mt < 4; ++mt)
#pragma unroll
      for (int nt = 0; nt < 2; ++nt)
        acc[mt][nt] = wmma_bf16(af[mt], bfg[nt], acc[mt][nt]);

    if (more) storeW(Bs[cur ^ 1]);          // loadcnt wait lands after compute
    wait_async0();                          // async A tile for kt+1 complete
    cur ^= 1;
  }

  // --- epilogue: bias + residual + relu, fp32/bf16 stores ---
#pragma unroll
  for (int nt = 0; nt < 2; ++nt) {
    const int col = bn + wn * 32 + nt * 16 + ln;
    if (col >= N) continue;
    const float bv = bias ? bias[col] : 0.f;
#pragma unroll
    for (int mt = 0; mt < 4; ++mt) {
#pragma unroll
      for (int r = 0; r < 8; ++r) {
        const int row = bm + wm * 64 + mt * 16 + halfIdx * 8 + r;
        float v = acc[mt][nt][r] + bv;
        if (residual) v += residual[(size_t)row * N + col];
        if (relu) v = fmaxf(v, 0.f);
        if (outF) outF[(size_t)row * N + col] = v;
        if (outB) outB[(size_t)row * N + col] = f2bf(v);
      }
    }
  }
}

// ---------------------------------------------------------------------------
// Causal flash attention. Block = one (b,h) x 128 queries, 8 waves x 16 rows.
// ---------------------------------------------------------------------------
__global__ __launch_bounds__(256) void attn_kernel(
    const __bf16* __restrict__ Q, const __bf16* __restrict__ Kb,
    const __bf16* __restrict__ Vb, __bf16* __restrict__ O) {
  const int qblock = blockIdx.x;            // T/128 tiles of queries
  const int bh     = blockIdx.y;            // b*H + h
  const int b      = bh >> 4;               // H = 16
  const int h      = bh & 15;
  const int tid    = threadIdx.x;
  const int lane   = tid & 31, wave = tid >> 5;
  const int ln     = lane & 15;
  const int halfIdx = lane >> 4;
  const int koff   = halfIdx * 8;

  __shared__ alignas(16) __bf16 Ks[32][80];        // [key][hs]   stride 160 B
  __shared__ alignas(16) __bf16 Vst[64][48];       // [hs][key]   stride  96 B
  __shared__ alignas(16) __bf16 Pw[8][16][32];     // per-wave P transpose

  const size_t rowbase = (size_t)b * TFX_T;
  const int qbase   = qblock * 128 + wave * 16;
  const int colbase = h * TFX_HS;

  // Q fragments (16 rows x 64 hs = two K=32 fragments), loaded once
  bf16x16 qf[2];
#pragma unroll
  for (int j = 0; j < 2; ++j)
    qf[j] = ld_frag(Q + (rowbase + qbase + ln) * TFX_C + colbase + j * 32, koff);

  float mrow[8], lrow[8];
  f32x8 of[4];
#pragma unroll
  for (int r = 0; r < 8; ++r) { mrow[r] = -3.0e38f; lrow[r] = 0.f; }
#pragma unroll
  for (int f = 0; f < 4; ++f)
#pragma unroll
    for (int i = 0; i < 8; ++i) of[f][i] = 0.f;

  const int nchunks = qblock * 4 + 4;       // keys needed: (qblock+1)*128
  for (int ch = 0; ch < nchunks; ++ch) {
    const int t0 = ch * 32;
    __syncthreads();
    // cooperative stage: K chunk via async DMA, V chunk transposed (paired)
    {
      const int kk = tid >> 3, c8 = (tid & 7) << 3;
      async_copy16(Kb + (rowbase + t0 + kk) * TFX_C + colbase + c8,
                   &Ks[kk][c8]);
      const int kp  = tid >> 4;             // key pair 0..15
      const int hs4 = (tid & 15) << 2;      // hs group 0..60
      bf16x4 v0 = *reinterpret_cast<const bf16x4*>(
          Vb + (rowbase + t0 + 2 * kp + 0) * TFX_C + colbase + hs4);
      bf16x4 v1 = *reinterpret_cast<const bf16x4*>(
          Vb + (rowbase + t0 + 2 * kp + 1) * TFX_C + colbase + hs4);
#pragma unroll
      for (int i = 0; i < 4; ++i) {
        bf16x2 pk; pk[0] = v0[i]; pk[1] = v1[i];
        *reinterpret_cast<bf16x2*>(&Vst[hs4 + i][2 * kp]) = pk;
      }
    }
    wait_async0();
    __syncthreads();

    if (t0 <= qbase + 15) {                 // chunk has at least one valid key
      // S = Q K^T over this 32-key chunk: two 16x16 tiles
      f32x8 s[2];
#pragma unroll
      for (int nt = 0; nt < 2; ++nt) {
#pragma unroll
        for (int i = 0; i < 8; ++i) s[nt][i] = 0.f;
        bf16x16 kf0 = ld_frag(&Ks[nt * 16 + ln][0], koff);   // hs 0..31
        bf16x16 kf1 = ld_frag(&Ks[nt * 16 + ln][32], koff);  // hs 32..63
        s[nt] = wmma_bf16(qf[0], kf0, s[nt]);
        s[nt] = wmma_bf16(qf[1], kf1, s[nt]);
      }
      const float scale = 0.125f;           // 1/sqrt(HS=64)
#pragma unroll
      for (int r = 0; r < 8; ++r) {
        const int qi = qbase + halfIdx * 8 + r;
        float v0 = s[0][r] * scale; if (t0 + ln > qi)      v0 = -3.0e38f;
        float v1 = s[1][r] * scale; if (t0 + 16 + ln > qi) v1 = -3.0e38f;
        float cmax = fmaxf(v0, v1);
#pragma unroll
        for (int off = 8; off; off >>= 1)
          cmax = fmaxf(cmax, __shfl_xor(cmax, off, 32));
        const float mnew = fmaxf(mrow[r], cmax);
        const float corr = __expf(mrow[r] - mnew);
        const float p0 = __expf(v0 - mnew);
        const float p1 = __expf(v1 - mnew);
        float ps = p0 + p1;
#pragma unroll
        for (int off = 8; off; off >>= 1) ps += __shfl_xor(ps, off, 32);
        lrow[r] = lrow[r] * corr + ps;
        mrow[r] = mnew;
#pragma unroll
        for (int f = 0; f < 4; ++f) of[f][r] *= corr;
        Pw[wave][halfIdx * 8 + r][ln]      = f2bf(p0);
        Pw[wave][halfIdx * 8 + r][16 + ln] = f2bf(p1);
      }
      // same-wave LDS RAW: lanes are lockstep, only need the DS counter
      asm volatile("s_wait_dscnt 0" ::: "memory");
      bf16x16 pf = ld_frag(&Pw[wave][ln][0], koff);
#pragma unroll
      for (int f = 0; f < 4; ++f) {
        bf16x16 vf = ld_frag(&Vst[f * 16 + ln][0], koff);
        of[f] = wmma_bf16(pf, vf, of[f]);
      }
    }
  }

  // normalize and store O (bf16, feeds the Wo GEMM)
#pragma unroll
  for (int f = 0; f < 4; ++f) {
    const int col = colbase + f * 16 + ln;
#pragma unroll
    for (int r = 0; r < 8; ++r) {
      const int row = qbase + halfIdx * 8 + r;
      O[(rowbase + row) * TFX_C + col] = f2bf(of[f][r] / lrow[r]);
    }
  }
}

// ---------------------------------------------------------------------------
// Launcher
// ---------------------------------------------------------------------------
extern "C" void kernel_launch(void* const* d_in, const int* in_sizes, int n_in,
                              void* d_out, int out_size, void* d_ws, size_t ws_size,
                              hipStream_t stream) {
  (void)in_sizes; (void)n_in; (void)out_size; (void)ws_size;
  const int*   idx  = (const int*)  d_in[0];
  const float* tok  = (const float*)d_in[1];
  const float* pos  = (const float*)d_in[2];
  const float* Wq   = (const float*)d_in[3];
  const float* Wk   = (const float*)d_in[4];
  const float* Wv   = (const float*)d_in[5];
  const float* Wo   = (const float*)d_in[6];
  const float* bo   = (const float*)d_in[7];
  const float* ln1g = (const float*)d_in[8];
  const float* ln1b = (const float*)d_in[9];
  const float* ln2g = (const float*)d_in[10];
  const float* ln2b = (const float*)d_in[11];
  const float* W1   = (const float*)d_in[12];
  const float* b1   = (const float*)d_in[13];
  const float* W2   = (const float*)d_in[14];
  const float* b2   = (const float*)d_in[15];
  const float* lnfg = (const float*)d_in[16];
  const float* lnfb = (const float*)d_in[17];
  const float* Wlm  = (const float*)d_in[18];
  const float* blm  = (const float*)d_in[19];

  const int M = TFX_B * TFX_T;               // 8192
  char* p = (char*)d_ws;
  float*  x   = (float*)p;   p += (size_t)M * TFX_C * sizeof(float);
  __bf16* h   = (__bf16*)p;  p += (size_t)M * TFX_C * 2;
  __bf16* qb  = (__bf16*)p;  p += (size_t)M * TFX_C * 2;
  __bf16* kb  = (__bf16*)p;  p += (size_t)M * TFX_C * 2;
  __bf16* vb  = (__bf16*)p;  p += (size_t)M * TFX_C * 2;
  __bf16* ob  = (__bf16*)p;  p += (size_t)M * TFX_C * 2;
  __bf16* ffh = (__bf16*)p;  p += (size_t)M * TFX_FF * 2;

  dim3 blk(256);
  dim3 gC(M / 128, TFX_C / 128);
  dim3 gF(M / 128, TFX_FF / 128);
  dim3 gV(M / 128, 1);
  dim3 gA(TFX_T / 128, TFX_B * TFX_H);

  embed_kernel<<<M, blk, 0, stream>>>(idx, tok, pos, x);
  for (int l = 0; l < TFX_L; ++l) {
    const size_t wofs = (size_t)l * TFX_C * TFX_C;
    ln_kernel<<<M, blk, 0, stream>>>(x, ln1g + l * TFX_C, ln1b + l * TFX_C, h);
    gemm_wmma<<<gC, blk, 0, stream>>>(h, Wq + wofs, nullptr, nullptr,
                                      nullptr, qb, M, TFX_C, TFX_C, 0);
    gemm_wmma<<<gC, blk, 0, stream>>>(h, Wk + wofs, nullptr, nullptr,
                                      nullptr, kb, M, TFX_C, TFX_C, 0);
    gemm_wmma<<<gC, blk, 0, stream>>>(h, Wv + wofs, nullptr, nullptr,
                                      nullptr, vb, M, TFX_C, TFX_C, 0);
    attn_kernel<<<gA, blk, 0, stream>>>(qb, kb, vb, ob);
    gemm_wmma<<<gC, blk, 0, stream>>>(ob, Wo + wofs, bo + l * TFX_C, x,
                                      x, nullptr, M, TFX_C, TFX_C, 0);
    ln_kernel<<<M, blk, 0, stream>>>(x, ln2g + l * TFX_C, ln2b + l * TFX_C, h);
    gemm_wmma<<<gF, blk, 0, stream>>>(h, W1 + (size_t)l * TFX_C * TFX_FF,
                                      b1 + l * TFX_FF, nullptr,
                                      nullptr, ffh, M, TFX_FF, TFX_C, 1);
    gemm_wmma<<<gC, blk, 0, stream>>>(ffh, W2 + (size_t)l * TFX_FF * TFX_C,
                                      b2 + l * TFX_C, x,
                                      x, nullptr, M, TFX_C, TFX_FF, 0);
  }
  ln_kernel<<<M, blk, 0, stream>>>(x, lnfg, lnfb, h);
  gemm_wmma<<<gV, blk, 0, stream>>>(h, Wlm, blm, nullptr,
                                    (float*)d_out, nullptr, M, TFX_V, TFX_C, 0);
}
